// TRANSGRAPH_80925773791326
// MI455X (gfx1250) — matrix operand order repaired and checked by reading, not compiled, not run on hardware
//
#include <hip/hip_runtime.h>
#include <cmath>
#include <stdint.h>

#define N_NODES 20000
#define N_EDGES 320000
#define HEADS   8
#define HDIM    64
#define CH      512
#define LAYERS  2

typedef __attribute__((ext_vector_type(16))) __bf16 v16bf;
typedef __attribute__((ext_vector_type(8)))  float  v8f;
typedef __attribute__((ext_vector_type(4)))  float  f32x4;
typedef __attribute__((ext_vector_type(4)))  int    i32x4;

union FragBF { v16bf v; f32x4 f[2]; };

#if defined(__AMDGCN__) && __has_builtin(__builtin_amdgcn_global_load_async_to_lds_b128)
#define USE_ASYNC_LDS 1
#endif

static __device__ __forceinline__ unsigned short f2bf(float x) {
  unsigned u = __float_as_uint(x);
  unsigned r = (u + 0x7FFFu + ((u >> 16) & 1u)) >> 16;   // round-to-nearest-even
  return (unsigned short)r;
}

// ---------------- utility kernels ----------------

__global__ __launch_bounds__(256) void fill_f32(float* __restrict__ p, float v, int n) {
  int i = blockIdx.x * blockDim.x + threadIdx.x;
  if (i < n) p[i] = v;
}

__global__ __launch_bounds__(256) void f32_to_bf16(const float* __restrict__ in,
                                                   unsigned short* __restrict__ out, int n) {
  int i = blockIdx.x * blockDim.x + threadIdx.x;
  if (i < n) out[i] = f2bf(in[i]);
}

// WT[n*K + k] = bf16(W[k*Nn + n])  (B-operand pre-transpose so WMMA B-frags are contiguous)
__global__ __launch_bounds__(256) void transpose_to_bf16(const float* __restrict__ W,
                                                         unsigned short* __restrict__ WT,
                                                         int K, int Nn) {
  int i = blockIdx.x * blockDim.x + threadIdx.x;
  if (i >= K * Nn) return;
  int n = i / K, k = i - n * K;
  WT[i] = f2bf(W[(size_t)k * Nn + n]);
}

__global__ __launch_bounds__(256) void add3(const float* __restrict__ a, const float* __restrict__ b,
                                            const float* __restrict__ c, float* __restrict__ o, int n) {
  int i = blockIdx.x * blockDim.x + threadIdx.x;
  if (i < n) o[i] = a[i] + b[i] + c[i];
}

// ---------------- bf16 WMMA GEMM: C[M,Nn] = A[M,K] @ B[K,Nn] (+bias,relu,residual) ----------------
// A row-major bf16; BT = B transposed, row-major bf16 [Nn,K].
// Block = 8 waves sharing one 16-row A tile (async-staged in LDS); each wave owns a 16x64 strip.
// REQUIREMENTS: grid exactly (M/16)*(Nn/64)/8 blocks; Nn/64 in {8,16}; K multiple of 64.

__global__ __launch_bounds__(256) void gemm_bf16(const unsigned short* __restrict__ A,
                                                 const unsigned short* __restrict__ BT,
                                                 const float* __restrict__ bias,
                                                 const float* __restrict__ residual,
                                                 float* __restrict__ C,
                                                 int M, int K, int Nn, int relu) {
  __shared__ unsigned short As[16 * (1024 + 8)];   // 16 rows, padded stride (+16B -> 4-bank shift/row)
  const int tid    = threadIdx.x;
  const int ngrp   = Nn >> 6;
  const int warp   = tid >> 5;
  const int wid    = blockIdx.x * 8 + warp;
  const int mt     = wid / ngrp;                   // block-uniform (ngrp in {8,16})
  const int ng     = wid - mt * ngrp;
  const int mt_blk = (blockIdx.x * 8) / ngrp;
  const int ldsK   = K + 8;

  // ---- stage A tile [16,K] into LDS (async DMA if available) ----
  const int chunksPerRow = K >> 3;                 // 16B chunks per row
  for (int c = tid; c < 16 * chunksPerRow; c += 256) {
    int row = c / chunksPerRow, col = c - row * chunksPerRow;
    const unsigned short* gp = A + (size_t)(mt_blk * 16 + row) * K + col * 8;
    unsigned short* lp = &As[row * ldsK + col * 8];
#ifdef USE_ASYNC_LDS
    __builtin_amdgcn_global_load_async_to_lds_b128(
        (__attribute__((address_space(1))) i32x4*)(void*)gp,
        (__attribute__((address_space(3))) i32x4*)(void*)lp, 0, 0);
#else
    *(f32x4*)lp = *(const f32x4*)gp;
#endif
  }
#ifdef USE_ASYNC_LDS
#if __has_builtin(__builtin_amdgcn_s_wait_asynccnt)
  __builtin_amdgcn_s_wait_asynccnt(0);
#else
  asm volatile("s_wait_asynccnt 0x0" ::: "memory");
#endif
#endif
  __syncthreads();

  const int lane   = tid & 31;
  const int half16 = lane & 15;
  const int g      = lane >> 4;
  const int n0     = ng * 64;
  const unsigned short* Al = As + half16 * ldsK;   // this wave's A rows live in LDS

  const unsigned short* Bb[4];
#pragma unroll
  for (int t = 0; t < 4; ++t)
    Bb[t] = BT + (size_t)(n0 + t * 16 + half16) * K + g * 8;

  v8f acc[4] = {};
  FragBF fa0, fa1, fb0[4], fb1[4];

  auto ldA = [&](int kb, FragBF& f) {              // ds_load_b128 x2; K = (h>=8?16:0)+8g+(h&7)
    f.f[0] = *(const f32x4*)(Al + kb + g * 8);
    f.f[1] = *(const f32x4*)(Al + kb + 16 + g * 8);
  };
  auto ldB = [&](int kb, FragBF* f) {              // global_load_b128 x8; K = (h>=8?16:0)+8g+(h&7)
#pragma unroll
    for (int t = 0; t < 4; ++t) {
      f[t].f[0] = *(const f32x4*)(Bb[t] + kb);
      f[t].f[1] = *(const f32x4*)(Bb[t] + kb + 16);
    }
  };
  auto mm = [&](FragBF& fa, FragBF* fb) {
#pragma unroll
    for (int t = 0; t < 4; ++t)
      acc[t] = __builtin_amdgcn_wmma_f32_16x16x32_bf16(false, fa.v, false, fb[t].v,
                                                       (short)0, acc[t], false, false);
  };

  // software pipeline: double-buffered fragments, loads for step i+1 overlap WMMAs of step i
  ldA(0, fa0); ldB(0, fb0);
  int kb = 0;
  for (; kb + 64 < K; kb += 64) {
    ldA(kb + 32, fa1); ldB(kb + 32, fb1);
    mm(fa0, fb0);
    ldA(kb + 64, fa0); ldB(kb + 64, fb0);
    mm(fa1, fb1);
  }
  ldA(kb + 32, fa1); ldB(kb + 32, fb1);
  mm(fa0, fb0);
  mm(fa1, fb1);

#pragma unroll
  for (int t = 0; t < 4; ++t) {
    int n = n0 + t * 16 + half16;
    float bv = bias ? bias[n] : 0.0f;
#pragma unroll
    for (int r = 0; r < 8; ++r) {                  // D layout: row = r + 8g, col = lane&15
      int row = mt * 16 + r + 8 * g;
      float val = acc[t][r] + bv;
      if (relu) val = val > 0.0f ? val : 0.0f;
      if (residual) val += residual[(size_t)row * Nn + n];
      C[(size_t)row * Nn + n] = val;
    }
  }
}

// ---------------- edge-phase kernels (memory-bound; f32x4 coalesced, fire-and-forget atomics) ----

// one wave per edge: a[e,h] = scale * dot(q[dst,h,:], k[src,h,:]); atomic segment-max into amax
__global__ __launch_bounds__(256) void edge_logits(const float* __restrict__ q,
                                                   const float* __restrict__ k,
                                                   const int* __restrict__ ei,
                                                   float* __restrict__ a,
                                                   float* __restrict__ amax, int E) {
  int e = blockIdx.x * (blockDim.x >> 5) + (threadIdx.x >> 5);
  if (e >= E) return;
  int lane = threadIdx.x & 31;
  int src = ei[e], dst = ei[E + e];
  const float* qr = q + (size_t)dst * CH;
  const float* kr = k + (size_t)src * CH;
  const float scale = 0.125f;  // 1/sqrt(64)
#pragma unroll
  for (int j = 0; j < 4; ++j) {
    int c = j * 128 + lane * 4;                   // lanes 0-15 -> head 2j, 16-31 -> head 2j+1
    f32x4 qv = *(const f32x4*)(qr + c);
    f32x4 kv = *(const f32x4*)(kr + c);
    float p = qv[0]*kv[0] + qv[1]*kv[1] + qv[2]*kv[2] + qv[3]*kv[3];
    p += __shfl_xor(p, 8, 32);
    p += __shfl_xor(p, 4, 32);
    p += __shfl_xor(p, 2, 32);
    p += __shfl_xor(p, 1, 32);
    if ((lane & 15) == 0) {
      int h = 2 * j + (lane >> 4);
      float logit = p * scale;
      a[(size_t)e * HEADS + h] = logit;
      float* addr = amax + (size_t)dst * HEADS + h;   // float atomic-max trick (init -inf)
      if (logit >= 0.0f) atomicMax((int*)addr, __float_as_int(logit));
      else               atomicMin((unsigned int*)addr, __float_as_uint(logit));
    }
  }
}

__global__ __launch_bounds__(256) void edge_exp(const int* __restrict__ ei,
                                                float* __restrict__ a,
                                                const float* __restrict__ amax,
                                                float* __restrict__ denom, int E) {
  int i = blockIdx.x * blockDim.x + threadIdx.x;
  if (i >= E * HEADS) return;
  int e = i >> 3, h = i & 7;
  int dst = ei[E + e];
  float v = expf(a[i] - amax[(size_t)dst * HEADS + h]);
  a[i] = v;
  atomicAdd(denom + (size_t)dst * HEADS + h, v);
}

// one wave per edge: agg[dst,:] += alpha[e,h(c)] * v[src,:]
__global__ __launch_bounds__(256) void edge_agg(const int* __restrict__ ei,
                                                const float* __restrict__ a,
                                                const float* __restrict__ denom,
                                                const float* __restrict__ v,
                                                float* __restrict__ agg, int E) {
  int e = blockIdx.x * (blockDim.x >> 5) + (threadIdx.x >> 5);
  if (e >= E) return;
  int lane = threadIdx.x & 31;
  int src = ei[e], dst = ei[E + e];
  const float* vr = v + (size_t)src * CH;
  float* ar = agg + (size_t)dst * CH;
#pragma unroll
  for (int j = 0; j < 4; ++j) {
    int c = j * 128 + lane * 4;
    int h = c >> 6;
    float alpha = a[(size_t)e * HEADS + h] / (denom[(size_t)dst * HEADS + h] + 1e-16f);
    f32x4 vv = *(const f32x4*)(vr + c);
    atomicAdd(ar + c + 0, alpha * vv[0]);
    atomicAdd(ar + c + 1, alpha * vv[1]);
    atomicAdd(ar + c + 2, alpha * vv[2]);
    atomicAdd(ar + c + 3, alpha * vv[3]);
  }
}

// one wave per row, single pass sum/sumsq + wave32 shfl reduction
__global__ __launch_bounds__(256) void layernorm_k(const float* __restrict__ X,
                                                   const float* __restrict__ gamma,
                                                   const float* __restrict__ beta,
                                                   const float* __restrict__ addv,
                                                   float* __restrict__ Y, int rows) {
  int row = blockIdx.x * (blockDim.x >> 5) + (threadIdx.x >> 5);
  if (row >= rows) return;
  int lane = threadIdx.x & 31;
  const float* xr = X + (size_t)row * CH;
  f32x4 vals[4];
  float s = 0.f, s2 = 0.f;
#pragma unroll
  for (int j = 0; j < 4; ++j) {
    int c = j * 128 + lane * 4;
    vals[j] = *(const f32x4*)(xr + c);
#pragma unroll
    for (int t = 0; t < 4; ++t) { s += vals[j][t]; s2 += vals[j][t] * vals[j][t]; }
  }
#pragma unroll
  for (int m = 16; m >= 1; m >>= 1) { s += __shfl_xor(s, m, 32); s2 += __shfl_xor(s2, m, 32); }
  float mu  = s * (1.0f / CH);
  float var = s2 * (1.0f / CH) - mu * mu;
  float rs  = rsqrtf(var + 1e-5f);
#pragma unroll
  for (int j = 0; j < 4; ++j) {
    int c = j * 128 + lane * 4;
    f32x4 o;
#pragma unroll
    for (int t = 0; t < 4; ++t) {
      float val = (vals[j][t] - mu) * rs * gamma[c + t] + beta[c + t];
      if (addv) val += addv[(size_t)row * CH + c + t];
      o[t] = val;
    }
    *(f32x4*)(Y + (size_t)row * CH + c) = o;
  }
}

// ---------------- driver ----------------

extern "C" void kernel_launch(void* const* d_in, const int* in_sizes, int n_in,
                              void* d_out, int out_size, void* d_ws, size_t ws_size,
                              hipStream_t stream) {
  const float* x   = (const float*)d_in[0];
  const int*   ei  = (const int*)d_in[1];
  // d_in[2] edge_type, d_in[3] edge_repre: unused by reference math
  const float* Wq  = (const float*)d_in[4];   const float* bq = (const float*)d_in[5];
  const float* Wk  = (const float*)d_in[6];   const float* bk = (const float*)d_in[7];
  const float* Wv  = (const float*)d_in[8];   const float* bv = (const float*)d_in[9];
  const float* Ws  = (const float*)d_in[10];  const float* bs = (const float*)d_in[11];
  const float* W1  = (const float*)d_in[12];  const float* b1 = (const float*)d_in[13];
  const float* W2  = (const float*)d_in[14];  const float* b2 = (const float*)d_in[15];
  const float* lng = (const float*)d_in[16];  const float* lnb = (const float*)d_in[17];

  const size_t NC = (size_t)N_NODES * CH;
  const int    cc = CH * CH;
  char* wsp = (char*)d_ws;
  auto carve = [&](size_t bytes) -> char* {
    char* p = wsp; wsp += (bytes + 255) & ~(size_t)255; return p;
  };
  float* xcur = (float*)carve(NC * 4);
  float* q    = (float*)carve(NC * 4);
  float* kk   = (float*)carve(NC * 4);     // q,kk contiguous -> reused as FFN hidden [N,2C]
  float* vvb  = (float*)carve(NC * 4);
  float* skip = (float*)carve(NC * 4);
  float* agg  = (float*)carve(NC * 4);     // reused as FFN output
  float* xln  = (float*)carve(NC * 4);
  float* alog = (float*)carve((size_t)N_EDGES * HEADS * 4);
  float* amax = (float*)carve((size_t)N_NODES * HEADS * 4);
  float* den  = (float*)carve((size_t)N_NODES * HEADS * 4);
  unsigned short* xb  = (unsigned short*)carve(NC * 2);
  unsigned short* hb  = (unsigned short*)carve(2 * NC * 2);
  unsigned short* WTq = (unsigned short*)carve((size_t)LAYERS * cc * 2);
  unsigned short* WTk = (unsigned short*)carve((size_t)LAYERS * cc * 2);
  unsigned short* WTv = (unsigned short*)carve((size_t)LAYERS * cc * 2);
  unsigned short* WTs = (unsigned short*)carve((size_t)LAYERS * cc * 2);
  unsigned short* WT1 = (unsigned short*)carve((size_t)2 * cc * 2);
  unsigned short* WT2 = (unsigned short*)carve((size_t)2 * cc * 2);
  float* h  = q;     // [N, 2C] spans q+kk
  float* ff = agg;   // FFN output

  // one-time (per call) weight convert + transpose to bf16
  for (int l = 0; l < LAYERS; ++l) {
    transpose_to_bf16<<<(cc + 255) / 256, 256, 0, stream>>>(Wq + (size_t)l * cc, WTq + (size_t)l * cc, CH, CH);
    transpose_to_bf16<<<(cc + 255) / 256, 256, 0, stream>>>(Wk + (size_t)l * cc, WTk + (size_t)l * cc, CH, CH);
    transpose_to_bf16<<<(cc + 255) / 256, 256, 0, stream>>>(Wv + (size_t)l * cc, WTv + (size_t)l * cc, CH, CH);
    transpose_to_bf16<<<(cc + 255) / 256, 256, 0, stream>>>(Ws + (size_t)l * cc, WTs + (size_t)l * cc, CH, CH);
  }
  transpose_to_bf16<<<(2 * cc + 255) / 256, 256, 0, stream>>>(W1, WT1, CH, 2 * CH);
  transpose_to_bf16<<<(2 * cc + 255) / 256, 256, 0, stream>>>(W2, WT2, 2 * CH, CH);

  (void)hipMemcpyAsync(xcur, x, NC * 4, hipMemcpyDeviceToDevice, stream);

  const int NH = N_NODES * HEADS;
  const int EH = N_EDGES * HEADS;
  const int gemmCC = ((N_NODES / 16) * (CH / 64)) / 8;       // exact: 1250
  const int gemmC2 = ((N_NODES / 16) * (2 * CH / 64)) / 8;   // exact: 2500

  for (int l = 0; l < LAYERS; ++l) {
    f32_to_bf16<<<((int)NC + 255) / 256, 256, 0, stream>>>(xcur, xb, (int)NC);
    gemm_bf16<<<gemmCC, 256, 0, stream>>>(xb, WTq + (size_t)l * cc, bq + l * CH, nullptr, q,    N_NODES, CH, CH, 0);
    gemm_bf16<<<gemmCC, 256, 0, stream>>>(xb, WTk + (size_t)l * cc, bk + l * CH, nullptr, kk,   N_NODES, CH, CH, 0);
    gemm_bf16<<<gemmCC, 256, 0, stream>>>(xb, WTv + (size_t)l * cc, bv + l * CH, nullptr, vvb,  N_NODES, CH, CH, 0);
    gemm_bf16<<<gemmCC, 256, 0, stream>>>(xb, WTs + (size_t)l * cc, bs + l * CH, nullptr, skip, N_NODES, CH, CH, 0);

    fill_f32<<<(NH + 255) / 256, 256, 0, stream>>>(amax, -INFINITY, NH);
    fill_f32<<<(NH + 255) / 256, 256, 0, stream>>>(den, 0.0f, NH);
    fill_f32<<<((int)NC + 255) / 256, 256, 0, stream>>>(agg, 0.0f, (int)NC);

    edge_logits<<<(N_EDGES + 7) / 8, 256, 0, stream>>>(q, kk, ei, alog, amax, N_EDGES);
    edge_exp<<<(EH + 255) / 256, 256, 0, stream>>>(ei, alog, amax, den, N_EDGES);
    edge_agg<<<(N_EDGES + 7) / 8, 256, 0, stream>>>(ei, alog, den, vvb, agg, N_EDGES);

    add3<<<((int)NC + 255) / 256, 256, 0, stream>>>(agg, skip, x, xcur, (int)NC);
  }

  // xln = LN(xcur)
  layernorm_k<<<(N_NODES + 7) / 8, 256, 0, stream>>>(xcur, lng, lnb, nullptr, xln, N_NODES);
  // h = relu(xln @ W1 + b1)
  f32_to_bf16<<<((int)NC + 255) / 256, 256, 0, stream>>>(xln, xb, (int)NC);
  gemm_bf16<<<gemmC2, 256, 0, stream>>>(xb, WT1, b1, nullptr, h, N_NODES, CH, 2 * CH, 1);
  // ff = h @ W2 + b2 + xln
  f32_to_bf16<<<((int)(2 * NC) + 255) / 256, 256, 0, stream>>>(h, hb, (int)(2 * NC));
  gemm_bf16<<<gemmCC, 256, 0, stream>>>(hb, WT2, b2, xln, ff, N_NODES, 2 * CH, CH, 0);
  // out = LN(ff) + enc
  layernorm_k<<<(N_NODES + 7) / 8, 256, 0, stream>>>(ff, lng, lnb, x, (float*)d_out, N_NODES);
}
